// Tagger_19988777796008
// MI455X (gfx1250) — matrix operand order repaired and checked by reading, not compiled, not run on hardware
//
#include <hip/hip_runtime.h>

#define D_VOCAB  10000
#define H_DIM    1024
#define M_ROWS   4096          // B*L = 8*512
#define N_LOGITS 10001         // D+1
#define WB_ROWS  10240         // full_vocab bf16, padded to grid coverage
#define WT_COLS  10016         // vocab^T bf16, K padded to 32-multiple

typedef __attribute__((ext_vector_type(16))) __bf16 v16bf;
typedef __attribute__((ext_vector_type(8)))  float  v8f;

// ---------------------------------------------------------------------------
// Async global->LDS 16-byte copy (CDNA5 GLOBAL_LOAD_ASYNC_TO_LDS_B128)
// Builtin signature (from hipcc diagnostic): param0 = v4i addrspace(1)*,
// then LDS pointer, imm offset, imm cpol.
// ---------------------------------------------------------------------------
__device__ __forceinline__ void async_cp16(void* lds, const void* g) {
#if defined(__HIP_DEVICE_COMPILE__)
#if __has_builtin(__builtin_amdgcn_global_load_async_to_lds_b128)
  typedef int v4i __attribute__((vector_size(16)));
  typedef __attribute__((address_space(1))) v4i* gas_t;
  typedef __attribute__((address_space(3))) v4i* las_t;
  __builtin_amdgcn_global_load_async_to_lds_b128((gas_t)g, (las_t)lds, 0, 0);
#else
  unsigned loff = (unsigned)(unsigned long long)
      (__attribute__((address_space(3))) void*)lds;
  unsigned long long ga = (unsigned long long)g;
  asm volatile("global_load_async_to_lds_b128 %0, %1, off"
               :: "v"(loff), "v"(ga) : "memory");
#endif
#endif
}

__device__ __forceinline__ void wait_async0() {
#if defined(__HIP_DEVICE_COMPILE__)
#if __has_builtin(__builtin_amdgcn_s_wait_asynccnt)
  __builtin_amdgcn_s_wait_asynccnt(0);
#else
  asm volatile("s_wait_asynccnt 0x0" ::: "memory");
#endif
#endif
}

// ---------------------------------------------------------------------------
// Prologue conversions (bandwidth-trivial: ~50 MB total @ 23.3 TB/s)
// ---------------------------------------------------------------------------
__global__ __launch_bounds__(256) void k_cvt_desc(
    const float* __restrict__ src, __bf16* __restrict__ dst)
{
  const size_t i0 = ((size_t)blockIdx.x * 256 + threadIdx.x) * 8;  // 4M elems
  #pragma unroll
  for (int i = 0; i < 8; ++i) dst[i0 + i] = (__bf16)src[i0 + i];
}

// Wb[r][c], r<10000: vocab; r==10000: default_embedding; r>10000: 0
__global__ __launch_bounds__(256) void k_cvt_wb(
    const float* __restrict__ vocab, const float* __restrict__ defe,
    __bf16* __restrict__ Wb)
{
  const size_t i0 = ((size_t)blockIdx.x * 256 + threadIdx.x) * 8;  // 10.49M elems
  const int r = (int)(i0 >> 10);
  const int c = (int)(i0 & 1023);
  if (r < D_VOCAB) {
    const float* p = &vocab[(size_t)r * H_DIM + c];
    #pragma unroll
    for (int i = 0; i < 8; ++i) Wb[i0 + i] = (__bf16)p[i];
  } else if (r == D_VOCAB) {
    #pragma unroll
    for (int i = 0; i < 8; ++i) Wb[i0 + i] = (__bf16)defe[c + i];
  } else {
    #pragma unroll
    for (int i = 0; i < 8; ++i) Wb[i0 + i] = (__bf16)0.f;
  }
}

// Wt[n][k] = vocab[k][n] (bf16), k padded 10000..10015 with 0. LDS tile transpose.
__global__ __launch_bounds__(256) void k_cvt_wt(
    const float* __restrict__ vocab, __bf16* __restrict__ Wt)
{
  __shared__ float tile[32][33];
  const int tid = threadIdx.x;
  const int k0 = blockIdx.x * 32;   // 313 tiles -> 10016
  const int n0 = blockIdx.y * 32;   // 32 tiles
  {
    const int lk = tid >> 3, ln4 = (tid & 7) * 4;
    float4 v = make_float4(0.f, 0.f, 0.f, 0.f);
    if (k0 + lk < D_VOCAB)
      v = *reinterpret_cast<const float4*>(&vocab[(size_t)(k0 + lk) * H_DIM + n0 + ln4]);
    tile[lk][ln4 + 0] = v.x; tile[lk][ln4 + 1] = v.y;
    tile[lk][ln4 + 2] = v.z; tile[lk][ln4 + 3] = v.w;
  }
  __syncthreads();
  {
    const int ln = tid >> 3, lk4 = (tid & 7) * 4;
    __bf16* p = &Wt[(size_t)(n0 + ln) * WT_COLS + k0 + lk4];
    #pragma unroll
    for (int i = 0; i < 4; ++i) p[i] = (__bf16)tile[lk4 + i][ln];
  }
}

// ---------------------------------------------------------------------------
// FAST Kernel 1: logits = Db(4096x1024) @ Wb(10240x1024)^T, bf16 inputs.
// Block 64(M) x 256(N), BK=32. 8 waves = 2(M) x 4(N); wave tile 32x64.
// Staging via async global->LDS b128 copies (no conversion in loop).
// ---------------------------------------------------------------------------
__global__ __launch_bounds__(256) void k_logits_fast(
    const __bf16* __restrict__ Db, const __bf16* __restrict__ Wb,
    float* __restrict__ sim)
{
  __shared__ __bf16 As[64][32];    //  4 KB
  __shared__ __bf16 Bs[256][32];   // 16 KB

  const int tid  = threadIdx.x;
  const int lane = tid & 31;
  const int wave = tid >> 5;
  const int wm   = wave >> 2;      // 0..1
  const int wn   = wave & 3;       // 0..3
  const int half = lane >> 4;
  const int l16  = lane & 15;

  const int n0 = blockIdx.x * 256;
  const int m0 = blockIdx.y * 64;

  v8f acc[2][4] = {};

  const int arow = tid >> 2;          // 0..63
  const int ach  = (tid & 3) * 8;     // bf16 elems, 16B chunks

  for (int k0 = 0; k0 < H_DIM; k0 += 32) {
    async_cp16(&As[arow][ach], Db + (size_t)(m0 + arow) * H_DIM + k0 + ach);
    #pragma unroll
    for (int j = 0; j < 4; ++j)
      async_cp16(&Bs[tid][j * 8], Wb + (size_t)(n0 + tid) * H_DIM + k0 + j * 8);
    wait_async0();
    __syncthreads();

    v16bf a[2];
    #pragma unroll
    for (int at = 0; at < 2; ++at) {
      const int am = wm * 32 + at * 16 + l16;
      #pragma unroll
      for (int i = 0; i < 8; ++i) {
        a[at][i]     = As[am][half * 8 + i];
        a[at][i + 8] = As[am][16 + half * 8 + i];
      }
    }
    #pragma unroll
    for (int nt = 0; nt < 4; ++nt) {
      v16bf b;
      const int bn = wn * 64 + nt * 16 + l16;
      #pragma unroll
      for (int i = 0; i < 8; ++i) { b[i] = Bs[bn][half * 8 + i]; b[i + 8] = Bs[bn][16 + half * 8 + i]; }
      #pragma unroll
      for (int at = 0; at < 2; ++at)
        acc[at][nt] = __builtin_amdgcn_wmma_f32_16x16x32_bf16(
            false, a[at], false, b, (short)0, acc[at][nt], false, false);
    }
    __syncthreads();
  }

  #pragma unroll
  for (int nt = 0; nt < 4; ++nt) {
    const int col = n0 + wn * 64 + nt * 16 + l16;
    if (col < N_LOGITS) {
      #pragma unroll
      for (int at = 0; at < 2; ++at) {
        #pragma unroll
        for (int r = 0; r < 8; ++r) {
          const int row = m0 + wm * 32 + at * 16 + r + 8 * half;
          sim[(size_t)row * N_LOGITS + col] = acc[at][nt][r];
        }
      }
    }
  }
}

// ---------------------------------------------------------------------------
// Kernel 2: in-place row softmax over 10001 columns (fp32, L2-resident)
// ---------------------------------------------------------------------------
__global__ __launch_bounds__(256) void k_softmax(float* __restrict__ sim)
{
  __shared__ float red[256];
  const int tid = threadIdx.x;
  float* row = sim + (size_t)blockIdx.x * N_LOGITS;

  float mx = -3.402823466e38f;
  for (int i = tid; i < N_LOGITS; i += 256) mx = fmaxf(mx, row[i]);
  red[tid] = mx; __syncthreads();
  for (int s = 128; s > 0; s >>= 1) {
    if (tid < s) red[tid] = fmaxf(red[tid], red[tid + s]);
    __syncthreads();
  }
  mx = red[0]; __syncthreads();

  float sum = 0.f;
  for (int i = tid; i < N_LOGITS; i += 256) {
    const float e = __expf(row[i] - mx);
    row[i] = e; sum += e;
  }
  red[tid] = sum; __syncthreads();
  for (int s = 128; s > 0; s >>= 1) {
    if (tid < s) red[tid] += red[tid + s];
    __syncthreads();
  }
  const float inv = 1.0f / red[0];
  __syncthreads();
  for (int i = tid; i < N_LOGITS; i += 256) row[i] *= inv;
}

// ---------------------------------------------------------------------------
// FAST Kernel 3: out = sim[:, :D] @ vocab + sim[:, D] * desc.
// B comes from pre-transposed bf16 Wt[n][k] via async copies; A (softmax
// weights) converted fp32->bf16 at staging. Block 64x128; wave tile 32x32.
// ---------------------------------------------------------------------------
__global__ __launch_bounds__(256) void k_blend_fast(
    const float* __restrict__ sim, const __bf16* __restrict__ Wt,
    const float* __restrict__ desc, float* __restrict__ out)
{
  __shared__ __bf16 As[64][32];    // 4 KB
  __shared__ __bf16 Bs[128][32];   // 8 KB

  const int tid  = threadIdx.x;
  const int lane = tid & 31;
  const int wave = tid >> 5;
  const int wm   = wave >> 2;
  const int wn   = wave & 3;
  const int half = lane >> 4;
  const int l16  = lane & 15;

  const int n0 = blockIdx.x * 128;
  const int m0 = blockIdx.y * 64;

  v8f acc[2][2] = {};

  for (int k0 = 0; k0 < WT_COLS; k0 += 32) {   // 313 iters
    // stage B from Wt (already padded with zeros past k=10000)
    {
      const int n = tid >> 1;
      #pragma unroll
      for (int j = 0; j < 2; ++j) {
        const int ch = (tid & 1) * 2 + j;
        async_cp16(&Bs[n][ch * 8],
                   Wt + (size_t)(n0 + n) * WT_COLS + k0 + ch * 8);
      }
    }
    // stage A: sim fp32 -> bf16 (10000 % 8 == 0, chunk-uniform guard)
    {
      const int row = tid >> 2;
      const int c8  = (tid & 3) * 8;
      const int kg  = k0 + c8;
      if (kg < D_VOCAB) {
        const float* p = &sim[(size_t)(m0 + row) * N_LOGITS + kg];
        #pragma unroll
        for (int i = 0; i < 8; ++i) As[row][c8 + i] = (__bf16)p[i];
      } else {
        #pragma unroll
        for (int i = 0; i < 8; ++i) As[row][c8 + i] = (__bf16)0.f;
      }
    }
    wait_async0();
    __syncthreads();

    v16bf a[2];
    #pragma unroll
    for (int at = 0; at < 2; ++at) {
      const int am = wm * 32 + at * 16 + l16;
      #pragma unroll
      for (int i = 0; i < 8; ++i) {
        a[at][i]     = As[am][half * 8 + i];
        a[at][i + 8] = As[am][16 + half * 8 + i];
      }
    }
    #pragma unroll
    for (int nt = 0; nt < 2; ++nt) {
      v16bf b;
      const int bn = wn * 32 + nt * 16 + l16;
      #pragma unroll
      for (int i = 0; i < 8; ++i) { b[i] = Bs[bn][half * 8 + i]; b[i + 8] = Bs[bn][16 + half * 8 + i]; }
      #pragma unroll
      for (int at = 0; at < 2; ++at)
        acc[at][nt] = __builtin_amdgcn_wmma_f32_16x16x32_bf16(
            false, a[at], false, b, (short)0, acc[at][nt], false, false);
    }
    __syncthreads();
  }

  #pragma unroll
  for (int nt = 0; nt < 2; ++nt) {
    const int col = n0 + wn * 32 + nt * 16 + l16;
    #pragma unroll
    for (int at = 0; at < 2; ++at) {
      #pragma unroll
      for (int r = 0; r < 8; ++r) {
        const int row = m0 + wm * 32 + at * 16 + r + 8 * half;
        const float w = sim[(size_t)row * N_LOGITS + D_VOCAB];
        out[(size_t)row * H_DIM + col] =
            acc[at][nt][r] + w * desc[(size_t)row * H_DIM + col];
      }
    }
  }
}

// ---------------------------------------------------------------------------
// FALLBACK kernels (convert-in-loop, no workspace needed)
// ---------------------------------------------------------------------------
__global__ __launch_bounds__(256) void k_logits_fb(
    const float* __restrict__ desc, const float* __restrict__ vocab,
    const float* __restrict__ defe, float* __restrict__ sim)
{
  __shared__ __bf16 As[32][32];
  __shared__ __bf16 Bs[256][32];

  const int tid  = threadIdx.x;
  const int lane = tid & 31;
  const int wave = tid >> 5;
  const int wm   = wave >> 2;
  const int wn   = wave & 3;
  const int half = lane >> 4;
  const int l16  = lane & 15;

  const int n0 = blockIdx.x * 256;
  const int m0 = blockIdx.y * 32;

  v8f acc[4] = {};
  const int srow = tid >> 3;
  const int sc4  = (tid & 7) * 4;

  for (int k0 = 0; k0 < H_DIM; k0 += 32) {
    {
      const float4 v = *reinterpret_cast<const float4*>(
          &desc[(size_t)(m0 + srow) * H_DIM + k0 + sc4]);
      As[srow][sc4 + 0] = (__bf16)v.x; As[srow][sc4 + 1] = (__bf16)v.y;
      As[srow][sc4 + 2] = (__bf16)v.z; As[srow][sc4 + 3] = (__bf16)v.w;
    }
    #pragma unroll
    for (int r = 0; r < 8; ++r) {
      const int row = srow + 32 * r;
      const int gn  = n0 + row;
      float4 v = make_float4(0.f, 0.f, 0.f, 0.f);
      if (gn < D_VOCAB)
        v = *reinterpret_cast<const float4*>(&vocab[(size_t)gn * H_DIM + k0 + sc4]);
      else if (gn == D_VOCAB)
        v = *reinterpret_cast<const float4*>(&defe[k0 + sc4]);
      Bs[row][sc4 + 0] = (__bf16)v.x; Bs[row][sc4 + 1] = (__bf16)v.y;
      Bs[row][sc4 + 2] = (__bf16)v.z; Bs[row][sc4 + 3] = (__bf16)v.w;
    }
    __syncthreads();

    v16bf a;
    const int am = wm * 16 + l16;
    #pragma unroll
    for (int i = 0; i < 8; ++i) {
      a[i]     = As[am][half * 8 + i];
      a[i + 8] = As[am][16 + half * 8 + i];
    }
    #pragma unroll
    for (int nt = 0; nt < 4; ++nt) {
      v16bf b;
      const int bn = wn * 64 + nt * 16 + l16;
      #pragma unroll
      for (int i = 0; i < 8; ++i) { b[i] = Bs[bn][half * 8 + i]; b[i + 8] = Bs[bn][16 + half * 8 + i]; }
      acc[nt] = __builtin_amdgcn_wmma_f32_16x16x32_bf16(
          false, a, false, b, (short)0, acc[nt], false, false);
    }
    __syncthreads();
  }

  #pragma unroll
  for (int nt = 0; nt < 4; ++nt) {
    const int col = n0 + wn * 64 + nt * 16 + l16;
    if (col < N_LOGITS) {
      #pragma unroll
      for (int r = 0; r < 8; ++r) {
        const int row = m0 + wm * 16 + r + 8 * half;
        sim[(size_t)row * N_LOGITS + col] = acc[nt][r];
      }
    }
  }
}

__global__ __launch_bounds__(256) void k_blend_fb(
    const float* __restrict__ sim, const float* __restrict__ vocab,
    const float* __restrict__ desc, float* __restrict__ out)
{
  __shared__ __bf16 As[32][32];
  __shared__ __bf16 Bs[128][32];

  const int tid  = threadIdx.x;
  const int lane = tid & 31;
  const int wave = tid >> 5;
  const int wm   = wave >> 2;
  const int wn   = wave & 3;
  const int half = lane >> 4;
  const int l16  = lane & 15;

  const int n0 = blockIdx.x * 128;
  const int m0 = blockIdx.y * 32;

  v8f acc[2] = {};

  for (int k0 = 0; k0 < WT_COLS; k0 += 32) {
    {
      const int row = tid >> 3;
      const int c4  = (tid & 7) * 4;
      const int kg  = k0 + c4;
      if (kg < D_VOCAB) {
        const float* p = &sim[(size_t)(m0 + row) * N_LOGITS + kg];
        As[row][c4 + 0] = (__bf16)p[0]; As[row][c4 + 1] = (__bf16)p[1];
        As[row][c4 + 2] = (__bf16)p[2]; As[row][c4 + 3] = (__bf16)p[3];
      } else {
        As[row][c4 + 0] = (__bf16)0.f; As[row][c4 + 1] = (__bf16)0.f;
        As[row][c4 + 2] = (__bf16)0.f; As[row][c4 + 3] = (__bf16)0.f;
      }
    }
    {
      const int kk  = tid >> 3;
      const int n16 = (tid & 7) * 16;
      const int kg  = k0 + kk;
      #pragma unroll
      for (int j = 0; j < 4; ++j) {
        const int n4 = n16 + j * 4;
        float4 v = make_float4(0.f, 0.f, 0.f, 0.f);
        if (kg < D_VOCAB)
          v = *reinterpret_cast<const float4*>(&vocab[(size_t)kg * H_DIM + n0 + n4]);
        Bs[n4 + 0][kk] = (__bf16)v.x; Bs[n4 + 1][kk] = (__bf16)v.y;
        Bs[n4 + 2][kk] = (__bf16)v.z; Bs[n4 + 3][kk] = (__bf16)v.w;
      }
    }
    __syncthreads();

    v16bf a;
    const int am = wm * 16 + l16;
    #pragma unroll
    for (int i = 0; i < 8; ++i) {
      a[i]     = As[am][half * 8 + i];
      a[i + 8] = As[am][16 + half * 8 + i];
    }
    #pragma unroll
    for (int nt = 0; nt < 2; ++nt) {
      v16bf b;
      const int bn = wn * 32 + nt * 16 + l16;
      #pragma unroll
      for (int i = 0; i < 8; ++i) { b[i] = Bs[bn][half * 8 + i]; b[i + 8] = Bs[bn][16 + half * 8 + i]; }
      acc[nt] = __builtin_amdgcn_wmma_f32_16x16x32_bf16(
          false, a, false, b, (short)0, acc[nt], false, false);
    }
    __syncthreads();
  }

  #pragma unroll
  for (int nt = 0; nt < 2; ++nt) {
    const int col = n0 + wn * 32 + nt * 16 + l16;
    #pragma unroll
    for (int r = 0; r < 8; ++r) {
      const int row = m0 + wm * 16 + r + 8 * half;
      const float w = sim[(size_t)row * N_LOGITS + D_VOCAB];
      out[(size_t)row * H_DIM + col] = acc[nt][r] + w * desc[(size_t)row * H_DIM + col];
    }
  }
}

// ---------------------------------------------------------------------------
extern "C" void kernel_launch(void* const* d_in, const int* in_sizes, int n_in,
                              void* d_out, int out_size, void* d_ws, size_t ws_size,
                              hipStream_t stream) {
  const float* vocab = (const float*)d_in[0];          // (10000, 1024)
  const float* desc  = (const float*)d_in[1];          // (8, 512, 1024)
  const float* defe  = (const float*)d_in[2];          // (1024,)
  float* out_cb  = (float*)d_out;                      // (8, 512, 1024)
  float* out_sim = out_cb + (size_t)M_ROWS * H_DIM;    // (8, 512, 10001)

  const size_t dbB = (size_t)M_ROWS * H_DIM * 2;       //  8,388,608
  const size_t wbB = (size_t)WB_ROWS * H_DIM * 2;      // 20,971,520
  const size_t wtB = (size_t)H_DIM * WT_COLS * 2;      // 20,512,768

  if (ws_size >= dbB + wbB + wtB) {
    __bf16* Db = (__bf16*)d_ws;
    __bf16* Wb = (__bf16*)((char*)d_ws + dbB);
    __bf16* Wt = (__bf16*)((char*)d_ws + dbB + wbB);

    k_cvt_desc<<<dim3(2048),     256, 0, stream>>>(desc, Db);
    k_cvt_wb  <<<dim3(5120),     256, 0, stream>>>(vocab, defe, Wb);
    k_cvt_wt  <<<dim3(313, 32),  256, 0, stream>>>(vocab, Wt);

    k_logits_fast<<<dim3(40, 64), 256, 0, stream>>>(Db, Wb, out_sim);
    k_softmax    <<<dim3(M_ROWS), 256, 0, stream>>>(out_sim);
    k_blend_fast <<<dim3(8, 64),  256, 0, stream>>>(out_sim, Wt, desc, out_cb);
  } else {
    k_logits_fb<<<dim3(40, 128), 256, 0, stream>>>(desc, vocab, defe, out_sim);
    k_softmax  <<<dim3(M_ROWS),  256, 0, stream>>>(out_sim);
    k_blend_fb <<<dim3(8, 128),  256, 0, stream>>>(out_sim, vocab, desc, out_cb);
  }
}
